// PerceiverAttention_10900626998040
// MI455X (gfx1250) — matrix operand order repaired and checked by reading, not compiled, not run on hardware
//
#include <hip/hip_runtime.h>
#include <hip/hip_bf16.h>
#include <stdint.h>

// ---------------------------------------------------------------------------
// PerceiverAttention for MI455X (gfx1250, wave32, WMMA bf16 16x16x32)
//  b=8, n=2048, nl=512, DIM=INNER=1024, heads=16, head_dim=64, keys=2560
// GEMMs: block = 4 waves x 32 rows (M=128, N=64); B tile staged in LDS via
//        global_load_async_to_lds_b128 (double-buffered, 1 barrier/k-step);
//        A fragments double-buffered in registers.
// ---------------------------------------------------------------------------

typedef __bf16 bf16;
typedef bf16  v8bf  __attribute__((ext_vector_type(8)));
typedef bf16  v16bf __attribute__((ext_vector_type(16)));
typedef float v8f   __attribute__((ext_vector_type(8)));
typedef int   v4i_  __attribute__((vector_size(16)));   // async-LDS payload type

#define B_    8
#define N_    2048
#define NL_   512
#define D_    1024
#define DH_   64
#define H_    16
#define NK_   (N_ + NL_)          // 2560
#define SCALE_   0.125f           // 64^-0.5
#define LN_EPS_  1e-5f
#define RMS_EPS_ 1e-8f
#define NEG_MAX_ (-3.402823466e+38f)

#define BS_STRIDE 40              // LDS row pitch (elements) for 32-elem rows
#define BS_BUF    (64 * BS_STRIDE)

#if defined(__has_builtin)
#if __has_builtin(__builtin_amdgcn_global_load_async_to_lds_b128) && \
    __has_builtin(__builtin_amdgcn_s_wait_asynccnt)
#define HAVE_ASYNC_LDS 1
#endif
#endif
#ifndef HAVE_ASYNC_LDS
#define HAVE_ASYNC_LDS 0
#endif

#if defined(__has_builtin) && __has_builtin(__builtin_amdgcn_wave_barrier)
#define WAVE_FENCE() __builtin_amdgcn_wave_barrier()
#else
#define WAVE_FENCE() __syncthreads()
#endif

static __device__ __forceinline__ v8f wmma_bf16(v16bf a, v16bf b, v8f c) {
  return __builtin_amdgcn_wmma_f32_16x16x32_bf16(false, a, false, b,
                                                 (short)0, c, false, false);
}

// A fragment (16x32 bf16 row-major, stride ld):
//   lane<16: K 0..7 + 16..23 ; lane>=16: K 8..15 + 24..31
static __device__ __forceinline__ v16bf load_a_frag(const bf16* tile, int ld) {
  const int lane = threadIdx.x & 31;
  const int hi   = lane >> 4;
  const bf16* p  = tile + (size_t)(lane & 15) * ld + hi * 8;
  v8bf c0 = *(const v8bf*)(p);
  v8bf c1 = *(const v8bf*)(p + 16);
  v16bf r;
#pragma unroll
  for (int i = 0; i < 8; ++i) { r[i] = c0[i]; r[i + 8] = c1[i]; }
  return r;
}

// B fragment from B^T row-major [n][k] (global), stride ld:
//   lane = column n; lane<16: K 0..7 + 16..23 ; lane>=16: K 8..15 + 24..31
static __device__ __forceinline__ v16bf load_b_frag(const bf16* tileT, int ld) {
  const int lane = threadIdx.x & 31;
  const int hi   = lane >> 4;
  const bf16* p  = tileT + (size_t)(lane & 15) * ld + hi * 8;
  v8bf c0 = *(const v8bf*)(p);
  v8bf c1 = *(const v8bf*)(p + 16);
  v16bf r;
#pragma unroll
  for (int i = 0; i < 8; ++i) { r[i] = c0[i]; r[i + 8] = c1[i]; }
  return r;
}

// B fragment from the LDS-staged tile (rows pitch BS_STRIDE)
static __device__ __forceinline__ v16bf lds_b_frag(const bf16* base) {
  const int lane = threadIdx.x & 31;
  const bf16* p  = base + (lane & 15) * BS_STRIDE + ((lane >> 4) << 3);
  v16bf r;
#pragma unroll
  for (int i = 0; i < 16; ++i) r[i] = p[i < 8 ? i : i + 8];
  return r;
}

// Stage one 64x32 bf16 B tile (B^T rows n0..n0+63, k-slice kt) into LDS.
// 128 threads x 2 chunks of 16B, via ASYNCcnt-tracked DMA when available.
static __device__ __forceinline__ void stage_B(const bf16* Bsrc, bf16* dst,
                                               int kt) {
  const int t = threadIdx.x;
#pragma unroll
  for (int i = 0; i < 2; ++i) {
    const int chunk = t * 2 + i;          // 0..255
    const int row   = chunk >> 2;         // 0..63
    const int co    = (chunk & 3) * 8;    // element offset in row
    const bf16* g = Bsrc + (size_t)row * D_ + kt + co;
    bf16* l = dst + row * BS_STRIDE + co;
#if HAVE_ASYNC_LDS
    __builtin_amdgcn_global_load_async_to_lds_b128((v4i_*)(void*)g,
                                                   (v4i_*)(void*)l, 0, 0);
#else
    *(v8bf*)l = *(const v8bf*)g;
#endif
  }
}

// Core: wave computes 32x64 tile; block computes 128x64.
// One barrier per k-step; B tile double-buffered in LDS, A fragments
// double-buffered in registers (loads issue one step ahead of use).
static __device__ __forceinline__ void gemm_tile_core(const bf16* Arow,
                                                      const bf16* Bcol,
                                                      bf16* Bs,
                                                      v8f c[2][4]) {
  stage_B(Bcol, Bs, 0);
  v16bf a0 = load_a_frag(Arow, D_);
  v16bf a1 = load_a_frag(Arow + (size_t)16 * D_, D_);
#pragma unroll 2
  for (int kt = 0; kt < D_; kt += 32) {
    const int cur = (kt >> 5) & 1;
#if HAVE_ASYNC_LDS
    __builtin_amdgcn_s_wait_asynccnt(0);
#endif
    __syncthreads();                               // buf[cur] ready, reads done
    int ktn = kt + 32;
    if (ktn >= D_) ktn = 0;                        // branch-free tail restage
    stage_B(Bcol, Bs + (cur ^ 1) * BS_BUF, ktn);
    // next-step A fragments: a full k-step of latency to cover them
    v16bf na0 = load_a_frag(Arow + ktn, D_);
    v16bf na1 = load_a_frag(Arow + (size_t)16 * D_ + ktn, D_);
    v16bf bfr[4];
#pragma unroll
    for (int f = 0; f < 4; ++f)
      bfr[f] = lds_b_frag(Bs + cur * BS_BUF + (f * 16) * BS_STRIDE);
#pragma unroll
    for (int f = 0; f < 4; ++f) {
      c[0][f] = wmma_bf16(a0, bfr[f], c[0][f]);
      c[1][f] = wmma_bf16(a1, bfr[f], c[1][f]);
    }
    a0 = na0;
    a1 = na1;
  }
}

// ---------------------------------------------------------------------------
// LayerNorm over last dim (1024) -> bf16.  One block (256 thr) per row.
// ---------------------------------------------------------------------------
__global__ void ln_bf16_kernel(const float* __restrict__ x,
                               const float* __restrict__ g,
                               const float* __restrict__ bta,
                               bf16* __restrict__ out) {
  __shared__ float s1[256];
  __shared__ float s2[256];
  const int row = blockIdx.x;
  const float* xr = x + (size_t)row * D_;
  float v[4];
  float sum = 0.f;
#pragma unroll
  for (int i = 0; i < 4; ++i) { v[i] = xr[threadIdx.x + i * 256]; sum += v[i]; }
  s1[threadIdx.x] = sum;
  __syncthreads();
  for (int off = 128; off > 0; off >>= 1) {
    if (threadIdx.x < off) s1[threadIdx.x] += s1[threadIdx.x + off];
    __syncthreads();
  }
  const float mu = s1[0] * (1.0f / D_);
  float ss = 0.f;
#pragma unroll
  for (int i = 0; i < 4; ++i) { float d = v[i] - mu; ss += d * d; }
  s2[threadIdx.x] = ss;
  __syncthreads();
  for (int off = 128; off > 0; off >>= 1) {
    if (threadIdx.x < off) s2[threadIdx.x] += s2[threadIdx.x + off];
    __syncthreads();
  }
  const float rstd = rsqrtf(s2[0] * (1.0f / D_) + LN_EPS_);
  bf16* orow = out + (size_t)row * D_;
#pragma unroll
  for (int i = 0; i < 4; ++i) {
    const int c = threadIdx.x + i * 256;
    orow[c] = (bf16)((v[i] - mu) * rstd * g[c] + bta[c]);
  }
}

// ---------------------------------------------------------------------------
// f32 [R][C] -> bf16 [C][R]  (weight transpose+convert), 32x32 LDS tiles
// ---------------------------------------------------------------------------
__global__ void transpose_cvt_kernel(const float* __restrict__ in,
                                     bf16* __restrict__ out, int R, int C) {
  __shared__ float tile[32][33];
  const int c0 = blockIdx.x * 32, r0 = blockIdx.y * 32;
  const int tx = threadIdx.x, ty = threadIdx.y;   // block (32,8)
#pragma unroll
  for (int i = 0; i < 4; ++i)
    tile[ty + i * 8][tx] = in[(size_t)(r0 + ty + i * 8) * C + c0 + tx];
  __syncthreads();
#pragma unroll
  for (int i = 0; i < 4; ++i)
    out[(size_t)(c0 + ty + i * 8) * R + r0 + tx] = (bf16)tile[tx][ty + i * 8];
}

// ---------------------------------------------------------------------------
// q = rmsnorm(LN(lat) @ W_q) * gamma * SCALE  -> qn[b][h][i][64] bf16
// grid (4096/128, 16 heads), block 128
// ---------------------------------------------------------------------------
__global__ void gemm_q_kernel(const bf16* __restrict__ A,    // lln [B*NL][D]
                              const bf16* __restrict__ BT,   // WqT [D][D]
                              const float* __restrict__ qg,  // [64]
                              bf16* __restrict__ qn) {
  __shared__ __align__(16) bf16 Bs[2 * BS_BUF];
  const int w    = threadIdx.x >> 5;
  const int lane = threadIdx.x & 31;
  const int m0w  = blockIdx.x * 128 + w * 32;   // wave's first row (b*NL+i)
  const int h    = blockIdx.y;
  v8f c[2][4] = {};
  gemm_tile_core(A + (size_t)m0w * D_, BT + (size_t)(h * DH_) * D_, Bs, c);

  const int hi = lane >> 4, col = lane & 15;
#pragma unroll
  for (int mf = 0; mf < 2; ++mf) {
    float inv[8];
#pragma unroll
    for (int r = 0; r < 8; ++r) {
      float ss = 0.f;
#pragma unroll
      for (int f = 0; f < 4; ++f) ss += c[mf][f][r] * c[mf][f][r];
#pragma unroll
      for (int m = 8; m >= 1; m >>= 1) ss += __shfl_xor(ss, m, 32);
      const float nrm = sqrtf(ss) * 0.125f;               // *64^-0.5
      inv[r] = (1.0f / fmaxf(nrm, RMS_EPS_)) * SCALE_;
    }
#pragma unroll
    for (int f = 0; f < 4; ++f) {
      const int d = f * 16 + col;
      const float gma = qg[d];
#pragma unroll
      for (int r = 0; r < 8; ++r) {
        const int row = m0w + mf * 16 + hi * 8 + r;
        const int b_ = row >> 9, i_ = row & 511;
        qn[((size_t)(b_ * H_ + h) * NL_ + i_) * DH_ + d] =
            (bf16)(c[mf][f][r] * inv[r] * gma);
      }
    }
  }
}

// ---------------------------------------------------------------------------
// kv = [xn;ln] @ W_kv ;  k -> rmsnorm*gamma -> kn[b][h][j][64]
//                        v -> transposed    -> vT[b][h][64][j]
// grid (B*NK/128=160, 2048/64=32), block 128
// ---------------------------------------------------------------------------
__global__ void gemm_kv_kernel(const bf16* __restrict__ xn,
                               const bf16* __restrict__ ln,
                               const bf16* __restrict__ BT,   // WkvT [2048][1024]
                               const float* __restrict__ kg,  // [64]
                               bf16* __restrict__ kn,
                               bf16* __restrict__ vT) {
  __shared__ __align__(16) bf16 Bs[2 * BS_BUF];
  const int w    = threadIdx.x >> 5;
  const int lane = threadIdx.x & 31;
  const int gr   = blockIdx.x * 128;          // block's first global row
  const int b_   = gr / NK_;
  const int j0   = gr % NK_;                  // block stays in one segment
  const int jw   = j0 + w * 32;               // wave's first key index
  const bf16* Arow = (j0 < N_)
      ? xn + ((size_t)b_ * N_ + jw) * D_
      : ln + ((size_t)b_ * NL_ + (jw - N_)) * D_;
  const int n0 = blockIdx.y * 64;
  v8f c[2][4] = {};
  gemm_tile_core(Arow, BT + (size_t)n0 * D_, Bs, c);

  const int hi = lane >> 4, col = lane & 15;
  if (n0 < D_) {                              // K region: rmsnorm*gamma
    const int h = n0 >> 6;
#pragma unroll
    for (int mf = 0; mf < 2; ++mf) {
      float inv[8];
#pragma unroll
      for (int r = 0; r < 8; ++r) {
        float ss = 0.f;
#pragma unroll
        for (int f = 0; f < 4; ++f) ss += c[mf][f][r] * c[mf][f][r];
#pragma unroll
        for (int m = 8; m >= 1; m >>= 1) ss += __shfl_xor(ss, m, 32);
        const float nrm = sqrtf(ss) * 0.125f;
        inv[r] = 1.0f / fmaxf(nrm, RMS_EPS_);
      }
#pragma unroll
      for (int f = 0; f < 4; ++f) {
        const int d = f * 16 + col;
        const float gma = kg[d];
#pragma unroll
        for (int r = 0; r < 8; ++r) {
          const int j = jw + mf * 16 + hi * 8 + r;
          kn[((size_t)(b_ * H_ + h) * NK_ + j) * DH_ + d] =
              (bf16)(c[mf][f][r] * inv[r] * gma);
        }
      }
    }
  } else {                                    // V region: store transposed
    const int h = (n0 - D_) >> 6;
#pragma unroll
    for (int mf = 0; mf < 2; ++mf) {
#pragma unroll
      for (int f = 0; f < 4; ++f) {
        const int d = f * 16 + col;
#pragma unroll
        for (int r = 0; r < 8; ++r) {
          const int j = jw + mf * 16 + hi * 8 + r;
          vT[((size_t)(b_ * H_ + h) * DH_ + d) * NK_ + j] = (bf16)c[mf][f][r];
        }
      }
    }
  }
}

// ---------------------------------------------------------------------------
// Flash attention per (b,h): 4 waves x 16 query rows, stream 32 keys/iter.
// K fragments double-buffered in registers; P transpose tile is per-wave so
// only a scheduling fence (no block barrier) is needed around it.
// ---------------------------------------------------------------------------
__global__ void attn_kernel(const bf16* __restrict__ qn,
                            const bf16* __restrict__ kn,
                            const bf16* __restrict__ vT,
                            const uint8_t* __restrict__ mask,  // bool [B][N]
                            bf16* __restrict__ ao) {           // [B][NL][D]
  __shared__ __align__(16) bf16 Pl[4][16][32];
  const int w    = threadIdx.x >> 5;
  const int lane = threadIdx.x & 31;
  const int hi   = lane >> 4, col = lane & 15;
  const int bh   = blockIdx.y;
  const int b_   = bh >> 4, h = bh & 15;
  const int q0   = blockIdx.x * 64 + w * 16;
  const bf16* Q = qn + ((size_t)bh * NL_ + q0) * DH_;
  const bf16* K = kn + (size_t)bh * NK_ * DH_;
  const bf16* V = vT + (size_t)bh * DH_ * NK_;
  const uint8_t* mrow = mask + (size_t)b_ * N_;

  const v16bf qa0 = load_a_frag(Q, DH_);
  const v16bf qa1 = load_a_frag(Q + 32, DH_);
  v8f o[4] = {};
  float mx[8], lsum[8];
#pragma unroll
  for (int r = 0; r < 8; ++r) { mx[r] = NEG_MAX_; lsum[r] = 0.f; }

  // current K-block fragments [jg][dchunk]
  v16bf kc[2][2];
#pragma unroll
  for (int jg = 0; jg < 2; ++jg) {
    kc[jg][0] = load_b_frag(K + (size_t)(jg * 16) * DH_, DH_);
    kc[jg][1] = load_b_frag(K + (size_t)(jg * 16) * DH_ + 32, DH_);
  }

  for (int kb = 0; kb < NK_; kb += 32) {
    // V fragments for this block (loads overlap the S WMMAs)
    v16bf vb[4];
#pragma unroll
    for (int f = 0; f < 4; ++f)
      vb[f] = load_b_frag(V + (size_t)(f * 16) * NK_ + kb, NK_);

    v8f s[2] = {};
#pragma unroll
    for (int jg = 0; jg < 2; ++jg) {
      s[jg] = wmma_bf16(qa0, kc[jg][0], s[jg]);
      s[jg] = wmma_bf16(qa1, kc[jg][1], s[jg]);
    }
    // preload next K block; overlaps the VALU-heavy softmax below
    if (kb + 32 < NK_) {
      const bf16* Kn = K + (size_t)(kb + 32) * DH_;
#pragma unroll
      for (int jg = 0; jg < 2; ++jg) {
        kc[jg][0] = load_b_frag(Kn + (size_t)(jg * 16) * DH_, DH_);
        kc[jg][1] = load_b_frag(Kn + (size_t)(jg * 16) * DH_ + 32, DH_);
      }
    }
    // mask (key j < N_ uses mask, latent keys always valid)
#pragma unroll
    for (int jg = 0; jg < 2; ++jg) {
      const int j = kb + jg * 16 + col;
      const bool valid = (j >= N_) || (mrow[j] != 0);
      if (!valid) {
#pragma unroll
        for (int r = 0; r < 8; ++r) s[jg][r] = NEG_MAX_;
      }
    }
    // online softmax (stats replicated across each 16-lane half group)
    float alpha[8];
#pragma unroll
    for (int r = 0; r < 8; ++r) {
      float m = fmaxf(s[0][r], s[1][r]);
#pragma unroll
      for (int mm = 8; mm >= 1; mm >>= 1) m = fmaxf(m, __shfl_xor(m, mm, 32));
      const float mnew = fmaxf(mx[r], m);
      alpha[r] = __expf(mx[r] - mnew);
      mx[r] = mnew;
      const float p0 = __expf(s[0][r] - mnew);
      const float p1 = __expf(s[1][r] - mnew);
      s[0][r] = p0; s[1][r] = p1;
      float ps = p0 + p1;
#pragma unroll
      for (int mm = 8; mm >= 1; mm >>= 1) ps += __shfl_xor(ps, mm, 32);
      lsum[r] = lsum[r] * alpha[r] + ps;
    }
    // P (C-layout) -> per-wave LDS tile -> A-layout fragment.
    // Same-wave LDS ops complete in order; no block barrier required.
#pragma unroll
    for (int r = 0; r < 8; ++r) {
      Pl[w][hi * 8 + r][col]      = (bf16)s[0][r];
      Pl[w][hi * 8 + r][16 + col] = (bf16)s[1][r];
    }
    WAVE_FENCE();
    v16bf pa;
#pragma unroll
    for (int i = 0; i < 8; ++i) {
      pa[i]     = Pl[w][col][hi * 8 + i];
      pa[i + 8] = Pl[w][col][hi * 8 + 16 + i];
    }
    WAVE_FENCE();
    // O = O*alpha + P @ Vblock
#pragma unroll
    for (int f = 0; f < 4; ++f) {
#pragma unroll
      for (int r = 0; r < 8; ++r) o[f][r] *= alpha[r];
      o[f] = wmma_bf16(pa, vb[f], o[f]);
    }
  }
  // normalize and merge heads
#pragma unroll
  for (int r = 0; r < 8; ++r) lsum[r] = 1.0f / lsum[r];
#pragma unroll
  for (int f = 0; f < 4; ++f) {
    const int d = f * 16 + col;
#pragma unroll
    for (int r = 0; r < 8; ++r) {
      const int qi = q0 + hi * 8 + r;
      ao[(size_t)(b_ * NL_ + qi) * D_ + h * DH_ + d] = (bf16)(o[f][r] * lsum[r]);
    }
  }
}

// ---------------------------------------------------------------------------
// out = attn_merged @ W_out + b_out  (f32 output)  grid (32, 16), block 128
// ---------------------------------------------------------------------------
__global__ void gemm_out_kernel(const bf16* __restrict__ A,    // ao [B*NL][D]
                                const bf16* __restrict__ BT,   // WoutT [D][D]
                                const float* __restrict__ bias,
                                float* __restrict__ out) {
  __shared__ __align__(16) bf16 Bs[2 * BS_BUF];
  const int w    = threadIdx.x >> 5;
  const int lane = threadIdx.x & 31;
  const int m0w  = blockIdx.x * 128 + w * 32;
  const int n0   = blockIdx.y * 64;
  v8f c[2][4] = {};
  gemm_tile_core(A + (size_t)m0w * D_, BT + (size_t)n0 * D_, Bs, c);

  const int hi = lane >> 4, col = lane & 15;
#pragma unroll
  for (int mf = 0; mf < 2; ++mf) {
#pragma unroll
    for (int f = 0; f < 4; ++f) {
      const int n = n0 + f * 16 + col;
      const float bv = bias[n];
#pragma unroll
      for (int r = 0; r < 8; ++r)
        out[(size_t)(m0w + mf * 16 + hi * 8 + r) * D_ + n] = c[mf][f][r] + bv;
    }
  }
}

// ---------------------------------------------------------------------------
extern "C" void kernel_launch(void* const* d_in, const int* in_sizes, int n_in,
                              void* d_out, int out_size, void* d_ws, size_t ws_size,
                              hipStream_t stream) {
  const float*   x    = (const float*)d_in[0];
  const float*   lat  = (const float*)d_in[1];
  const uint8_t* mask = (const uint8_t*)d_in[2];   // jnp bool -> 1 byte/elem
  const float*   lnxg = (const float*)d_in[3];
  const float*   lnxb = (const float*)d_in[4];
  const float*   lnlg = (const float*)d_in[5];
  const float*   lnlb = (const float*)d_in[6];
  const float*   qg   = (const float*)d_in[7];
  const float*   kg   = (const float*)d_in[8];
  const float*   Wq   = (const float*)d_in[9];
  const float*   Wkv  = (const float*)d_in[10];
  const float*   Wout = (const float*)d_in[11];
  const float*   bout = (const float*)d_in[12];
  float* out = (float*)d_out;

  uint8_t* ws = (uint8_t*)d_ws;
  size_t off = 0;
  auto alloc = [&](size_t bytes) -> void* {
    void* p = ws + off;
    off = (off + bytes + 255) & ~(size_t)255;
    return p;
  };
  bf16* xln  = (bf16*)alloc((size_t)B_ * N_  * D_ * 2);        // 32 MB
  bf16* lln  = (bf16*)alloc((size_t)B_ * NL_ * D_ * 2);        //  8 MB
  bf16* WqT  = (bf16*)alloc((size_t)D_ * D_ * 2);              //  2 MB
  bf16* WkvT = (bf16*)alloc((size_t)2 * D_ * D_ * 2);          //  4 MB
  bf16* WoT  = (bf16*)alloc((size_t)D_ * D_ * 2);              //  2 MB
  bf16* qn   = (bf16*)alloc((size_t)B_ * H_ * NL_ * DH_ * 2);  //  8 MB
  bf16* kn   = (bf16*)alloc((size_t)B_ * H_ * NK_ * DH_ * 2);  // 40 MB
  bf16* vT   = (bf16*)alloc((size_t)B_ * H_ * DH_ * NK_ * 2);  // 40 MB
  bf16* ao   = (bf16*)alloc((size_t)B_ * NL_ * D_ * 2);        //  8 MB

  ln_bf16_kernel<<<B_ * N_,  256, 0, stream>>>(x,   lnxg, lnxb, xln);
  ln_bf16_kernel<<<B_ * NL_, 256, 0, stream>>>(lat, lnlg, lnlb, lln);
  transpose_cvt_kernel<<<dim3(D_ / 32,     D_ / 32), dim3(32, 8), 0, stream>>>(Wq,   WqT,  D_, D_);
  transpose_cvt_kernel<<<dim3(2 * D_ / 32, D_ / 32), dim3(32, 8), 0, stream>>>(Wkv,  WkvT, D_, 2 * D_);
  transpose_cvt_kernel<<<dim3(D_ / 32,     D_ / 32), dim3(32, 8), 0, stream>>>(Wout, WoT,  D_, D_);
  gemm_q_kernel  <<<dim3(B_ * NL_ / 128, H_),          128, 0, stream>>>(lln, WqT, qg, qn);
  gemm_kv_kernel <<<dim3(B_ * NK_ / 128, 2 * D_ / 64), 128, 0, stream>>>(xln, lln, WkvT, kg, kn, vT);
  attn_kernel    <<<dim3(NL_ / 64, B_ * H_),           128, 0, stream>>>(qn, kn, vT, mask, ao);
  gemm_out_kernel<<<dim3(B_ * NL_ / 128, D_ / 64),     128, 0, stream>>>(ao, WoT, bout, out);
}
